// RepeatMixer_59554016526927
// MI455X (gfx1250) — matrix operand order repaired and checked by reading, not compiled, not run on hardware
//
#include <hip/hip_runtime.h>
#include <hip/hip_bf16.h>

typedef __attribute__((ext_vector_type(16))) _Float16 v16h;
typedef __attribute__((ext_vector_type(8)))  _Float16 v8h;
typedef __attribute__((ext_vector_type(8)))  float    v8f;

#define NB   2048   // batch
#define NK   20     // tokens per side
#define ND   128    // edge feature dim
#define NT   100    // time enc dim
#define NC   128    // channels
#define NTK  40     // 2K tokens
#define NHC  512    // channel hidden
#define HALF_ROWS 40960  // B*K rows per side

__device__ __forceinline__ float gelu_f(float x) {
    return 0.5f * x * (1.0f + erff(x * 0.70710678118654752440f));
}

// A fragment from a row-major f16 tile (LDS or global).
// lane l: row = l%16 ; K elements {kbase+off .. +7, kbase+off+16 .. +23}, off = 8*(l>=16)
__device__ __forceinline__ v16h load_a_frag(const _Float16* base, int stride_h, int kbase, int lane) {
    int m   = lane & 15;
    int off = (lane < 16) ? 0 : 8;
    const _Float16* p = base + (size_t)m * stride_h + kbase + off;
    v8h lo = *(const v8h*)p;
    v8h hi = *(const v8h*)(p + 16);
    v16h a;
#pragma unroll
    for (int e = 0; e < 8; ++e) { a[e] = lo[e]; a[e + 8] = hi[e]; }
    return a;
}

// B fragment from an N-major (pre-transposed) f16 weight: wt[n][k], row stride = Kp.
// lane l: col = n0 + l%16 ; K elements {kbase+off .. +7, kbase+off+16 .. +23}, off = 8*(l>=16)
__device__ __forceinline__ v16h load_b_frag(const _Float16* wt, int Kp, int n0, int kbase, int lane) {
    int col = n0 + (lane & 15);
    int off = (lane < 16) ? 0 : 8;
    const _Float16* p = wt + (size_t)col * Kp + kbase + off;
    v8h lo = *(const v8h*)p;
    v8h hi = *(const v8h*)(p + 16);
    v16h b;
#pragma unroll
    for (int e = 0; e < 8; ++e) { b[e] = lo[e]; b[e + 8] = hi[e]; }
    return b;
}

// ---------------- weight convert: f32 [K][N] -> f16 transposed padded [N][Kp] ----------------
__global__ void wconv_kernel(const float* __restrict__ src, _Float16* __restrict__ dst,
                             int K, int N, int Kp) {
    int i = blockIdx.x * blockDim.x + threadIdx.x;
    if (i >= N * Kp) return;
    int n = i / Kp, k = i - n * Kp;
    float v = (k < K) ? src[(size_t)k * N + n] : 0.0f;
    dst[i] = (_Float16)v;
}

// ---------------- one-hop token build: gather + time-enc + WMMA projection ----------------
#define K1P 256   // padded K for weights (228 -> 256)
#define A1S 264   // LDS stride (halves)
__global__ __launch_bounds__(256) void build_tokens1(
    const float* __restrict__ edge_table,
    const int*  __restrict__ src_eids, const int* __restrict__ dst_eids,
    const float* __restrict__ dt_src,  const float* __restrict__ dt_dst,
    const float* __restrict__ time_w,  const float* __restrict__ time_b,
    const _Float16* __restrict__ projWt, const float* __restrict__ proj_b,
    float* __restrict__ tokens)
{
    __shared__ _Float16 At[16 * A1S];
    const int tile = blockIdx.x;      // 16 rows of 81920 (side-major ordering)
    const int tid  = threadIdx.x;

    for (int idx = tid; idx < 16 * A1S; idx += 256) {
        int r = idx / A1S, c = idx - r * A1S;
        int gr   = tile * 16 + r;
        int side = (gr >= HALF_ROWS) ? 1 : 0;
        int rem  = gr - side * HALF_ROWS;
        int b = rem / NK, k = rem - b * NK;
        int eid = side ? dst_eids[b * NK + k] : src_eids[b * NK + k];
        float val = 0.0f;
        if (c < ND) {
            val = edge_table[(size_t)eid * ND + c];
        } else if (c < ND + NT) {
            if (eid != 0) {
                float dt = side ? dt_dst[b * NK + k] : dt_src[b * NK + k];
                int j = c - ND;
                val = cosf(dt * time_w[j] + time_b[j]);
            }
        }
        At[idx] = (_Float16)val;
    }
    __syncthreads();

    const int wave = tid >> 5, lane = tid & 31;
    v8f acc; 
#pragma unroll
    for (int e = 0; e < 8; ++e) acc[e] = 0.0f;
    const int n0 = wave * 16;
#pragma unroll
    for (int kk = 0; kk < K1P; kk += 32) {
        v16h a = load_a_frag(At, A1S, kk, lane);
        v16h b = load_b_frag(projWt, K1P, n0, kk, lane);
        acc = __builtin_amdgcn_wmma_f32_16x16x32_f16(false, a, false, b, (short)0, acc, false, false);
    }
    const int n = n0 + (lane & 15);
    const float bias = proj_b[n];
    const int mb = (lane < 16) ? 0 : 8;
#pragma unroll
    for (int v = 0; v < 8; ++v) {
        int gr   = tile * 16 + mb + v;
        int side = (gr >= HALF_ROWS) ? 1 : 0;
        int rem  = gr - side * HALF_ROWS;
        int b = rem / NK, k = rem - b * NK;
        tokens[((size_t)b * NTK + side * NK + k) * NC + n] = acc[v] + bias;
    }
}

// ---------------- two-hop token build (3 gathers + time-enc, per-side weights) ----------------
#define K2P 512   // padded K for weights (484 -> 512)
#define A2S 520   // LDS stride (halves)
__global__ __launch_bounds__(256) void build_tokens2(
    const float* __restrict__ edge_table,
    const int* __restrict__ src_eids,  const int* __restrict__ dst_eids,
    const int* __restrict__ src2_e1,   const int* __restrict__ src2_e2,
    const int* __restrict__ dst2_e1,   const int* __restrict__ dst2_e2,
    const float* __restrict__ dt2_src, const float* __restrict__ dt2_dst,
    const float* __restrict__ time_w,  const float* __restrict__ time_b,
    const _Float16* __restrict__ eprojWt, const float* __restrict__ eproj_b,
    float* __restrict__ tokens2)
{
    __shared__ _Float16 At[16 * A2S];
    const int tile = blockIdx.x;
    const int tid  = threadIdx.x;
    const int side_t = (tile * 16) / HALF_ROWS;   // uniform per tile (HALF_ROWS % 16 == 0)

    for (int idx = tid; idx < 16 * A2S; idx += 256) {
        int r = idx / A2S, c = idx - r * A2S;
        int gr  = tile * 16 + r;
        int rem = gr - side_t * HALF_ROWS;
        int b = rem / NK, k = rem - b * NK;
        int e1  = side_t ? dst2_e1[b * NK + k] : src2_e1[b * NK + k];
        float val = 0.0f;
        if (c < ND) {
            val = edge_table[(size_t)e1 * ND + c];
        } else if (c < 2 * ND) {
            int e2 = side_t ? dst2_e2[b * NK + k] : src2_e2[b * NK + k];
            val = edge_table[(size_t)e2 * ND + (c - ND)];
        } else if (c < 3 * ND) {
            int e0 = side_t ? dst_eids[b * NK + k] : src_eids[b * NK + k];
            val = edge_table[(size_t)e0 * ND + (c - 2 * ND)];
        } else if (c < 3 * ND + NT) {
            if (e1 != 0) {
                float dt = side_t ? dt2_dst[b * NK + k] : dt2_src[b * NK + k];
                int j = c - 3 * ND;
                val = cosf(dt * time_w[j] + time_b[j]);
            }
        }
        At[idx] = (_Float16)val;
    }
    __syncthreads();

    const int wave = tid >> 5, lane = tid & 31;
    const _Float16* W = eprojWt + (size_t)side_t * NC * K2P;
    v8f acc;
#pragma unroll
    for (int e = 0; e < 8; ++e) acc[e] = 0.0f;
    const int n0 = wave * 16;
#pragma unroll
    for (int kk = 0; kk < K2P; kk += 32) {
        v16h a = load_a_frag(At, A2S, kk, lane);
        v16h b = load_b_frag(W, K2P, n0, kk, lane);
        acc = __builtin_amdgcn_wmma_f32_16x16x32_f16(false, a, false, b, (short)0, acc, false, false);
    }
    const int n = n0 + (lane & 15);
    const float bias = eproj_b[side_t * NC + n];
    const int mb = (lane < 16) ? 0 : 8;
#pragma unroll
    for (int v = 0; v < 8; ++v) {
        int gr  = tile * 16 + mb + v;
        int rem = gr - side_t * HALF_ROWS;
        int b = rem / NK, k = rem - b * NK;
        tokens2[((size_t)b * NTK + side_t * NK + k) * NC + n] = acc[v] + bias;
    }
}

// ---------------- fused 2-layer mixer + token mean, one sample per workgroup ----------------
#define XST 130   // xs stride (dwords)
#define AST 136   // aN / g1 stride (halves)
__global__ __launch_bounds__(256) void mixer2_kernel(
    const float* __restrict__ x_in, float* __restrict__ emb_out, int layer_base,
    const float* __restrict__ tln_g, const float* __restrict__ tln_b,
    const float* __restrict__ tW1,   const float* __restrict__ tb1,
    const float* __restrict__ tW2,   const float* __restrict__ tb2,
    const float* __restrict__ cln_g, const float* __restrict__ cln_b,
    const _Float16* __restrict__ cW1t, const float* __restrict__ cb1,
    const _Float16* __restrict__ cW2t, const float* __restrict__ cb2)
{
    __shared__ float    xs[NTK * XST];       // residual stream, f32
    __shared__ _Float16 aN[48 * AST];        // LN'd input (WMMA A, rows 40..47 zero)
    __shared__ _Float16 g1[48 * AST];        // gelu(hidden) chunk (WMMA A for GEMM2)
    __shared__ float s_tg[NTK], s_tb[NTK], s_w1[NTK * 20], s_b1[20], s_w2[20 * NTK], s_b2[NTK];
    __shared__ float s_cg[NC], s_cb[NC];

    const int b = blockIdx.x, tid = threadIdx.x;
    const int wave = tid >> 5, lane = tid & 31;

    for (int i = tid; i < NTK * NC; i += 256) {
        int t = i >> 7, c = i & 127;
        xs[t * XST + c] = x_in[(size_t)b * NTK * NC + i];
    }
    for (int i = tid; i < 8 * AST; i += 256) aN[40 * AST + i] = (_Float16)0.0f;  // pad rows
    __syncthreads();

    for (int li = 0; li < 2; ++li) {
        const int L = layer_base + li;
        for (int i = tid; i < NTK; i += 256) {
            s_tg[i] = tln_g[L * NTK + i]; s_tb[i] = tln_b[L * NTK + i]; s_b2[i] = tb2[L * NTK + i];
        }
        for (int i = tid; i < NTK * 20; i += 256) { s_w1[i] = tW1[L * 800 + i]; s_w2[i] = tW2[L * 800 + i]; }
        if (tid < 20) s_b1[tid] = tb1[L * 20 + tid];
        if (tid < NC) { s_cg[tid] = cln_g[L * NC + tid]; s_cb[tid] = cln_b[L * NC + tid]; }
        __syncthreads();

        // ---- token mixing (tiny 40->20->40 FFN per channel, VALU) ----
        if (tid < NC) {
            const int c = tid;
            float m = 0.0f;
            for (int t = 0; t < NTK; ++t) m += xs[t * XST + c];
            m *= (1.0f / NTK);
            float var = 0.0f;
            for (int t = 0; t < NTK; ++t) { float d = xs[t * XST + c] - m; var += d * d; }
            var *= (1.0f / NTK);
            float inv = rsqrtf(var + 1e-5f);
            float hn[NTK];
            for (int t = 0; t < NTK; ++t) hn[t] = (xs[t * XST + c] - m) * inv * s_tg[t] + s_tb[t];
            float g[20];
            for (int j = 0; j < 20; ++j) {
                float a = s_b1[j];
                for (int t = 0; t < NTK; ++t) a += hn[t] * s_w1[t * 20 + j];
                g[j] = gelu_f(a);
            }
            for (int t = 0; t < NTK; ++t) {
                float a = s_b2[t];
                for (int j = 0; j < 20; ++j) a += g[j] * s_w2[j * NTK + t];
                xs[t * XST + c] += a;    // residual
            }
        }
        __syncthreads();

        // ---- channel LN -> f16 A matrix ----
        if (tid < NTK) {
            const int t = tid;
            float m = 0.0f;
            for (int c = 0; c < NC; ++c) m += xs[t * XST + c];
            m *= (1.0f / NC);
            float var = 0.0f;
            for (int c = 0; c < NC; ++c) { float d = xs[t * XST + c] - m; var += d * d; }
            var *= (1.0f / NC);
            float inv = rsqrtf(var + 1e-5f);
            for (int c = 0; c < NC; ++c)
                aN[t * AST + c] = (_Float16)((xs[t * XST + c] - m) * inv * s_cg[c] + s_cb[c]);
        }
        __syncthreads();

        // ---- channel FFN 128 -> 512 -> 128 via WMMA, hidden in 4 chunks of 128 ----
        const _Float16* W1 = cW1t + (size_t)L * NHC * NC;   // [512][128]
        const _Float16* W2 = cW2t + (size_t)L * NC * NHC;   // [128][512]
        v8f o[3];
#pragma unroll
        for (int j = 0; j < 3; ++j)
#pragma unroll
            for (int e = 0; e < 8; ++e) o[j][e] = 0.0f;

        for (int q = 0; q < 4; ++q) {
            // GEMM1: 3 M-tiles x 8 N-tiles = 24 tiles over 8 waves
#pragma unroll
            for (int j = 0; j < 3; ++j) {
                int tt = wave + 8 * j;
                int mt = tt % 3, nt = tt / 3;
                v8f acc;
#pragma unroll
                for (int e = 0; e < 8; ++e) acc[e] = 0.0f;
#pragma unroll
                for (int kk = 0; kk < NC; kk += 32) {
                    v16h a  = load_a_frag(aN + mt * 16 * AST, AST, kk, lane);
                    v16h bb = load_b_frag(W1, NC, q * 128 + nt * 16, kk, lane);
                    acc = __builtin_amdgcn_wmma_f32_16x16x32_f16(false, a, false, bb, (short)0, acc, false, false);
                }
                int n_in = nt * 16 + (lane & 15);
                float bias = cb1[L * NHC + q * 128 + n_in];
                int mb = (lane < 16) ? 0 : 8;
#pragma unroll
                for (int v = 0; v < 8; ++v)
                    g1[(mt * 16 + mb + v) * AST + n_in] = (_Float16)gelu_f(acc[v] + bias);
            }
            __syncthreads();
            // GEMM2 partial over this hidden chunk: accumulate output tiles in registers
#pragma unroll
            for (int j = 0; j < 3; ++j) {
                int tt = wave + 8 * j;
                int mt = tt % 3, nt = tt / 3;
#pragma unroll
                for (int kk = 0; kk < 128; kk += 32) {
                    v16h a  = load_a_frag(g1 + mt * 16 * AST, AST, kk, lane);
                    v16h bb = load_b_frag(W2, NHC, nt * 16, q * 128 + kk, lane);
                    o[j] = __builtin_amdgcn_wmma_f32_16x16x32_f16(false, a, false, bb, (short)0, o[j], false, false);
                }
            }
            __syncthreads();
        }
        // residual writeback
#pragma unroll
        for (int j = 0; j < 3; ++j) {
            int tt = wave + 8 * j;
            int mt = tt % 3, nt = tt / 3;
            int n  = nt * 16 + (lane & 15);
            float bias = cb2[L * NC + n];
            int mb = (lane < 16) ? 0 : 8;
#pragma unroll
            for (int v = 0; v < 8; ++v) {
                int grow = mt * 16 + mb + v;
                if (grow < NTK) xs[grow * XST + n] += o[j][v] + bias;
            }
        }
        __syncthreads();
    }

    // branch embedding: mean over tokens
    if (tid < NC) {
        float s = 0.0f;
        for (int t = 0; t < NTK; ++t) s += xs[t * XST + tid];
        emb_out[(size_t)b * NC + tid] = s * (1.0f / NTK);
    }
}

// ---------------- pcc-softmax combine ----------------
__global__ void combine_kernel(const float* __restrict__ edge_emb, const float* __restrict__ high_emb,
                               const float* __restrict__ pcc1, const float* __restrict__ pcc2,
                               float* __restrict__ out) {
    int b = blockIdx.x, c = threadIdx.x;
    float a = pcc1[b], d = pcc2[b];
    float mx = fmaxf(a, d);
    float e1 = expf(a - mx), e2 = expf(d - mx);
    float inv = 1.0f / (e1 + e2);
    out[(size_t)b * NC + c] = (e1 * inv) * edge_emb[(size_t)b * NC + c]
                            + (e2 * inv) * high_emb[(size_t)b * NC + c];
}

extern "C" void kernel_launch(void* const* d_in, const int* in_sizes, int n_in,
                              void* d_out, int out_size, void* d_ws, size_t ws_size,
                              hipStream_t stream) {
    const float* edge_table = (const float*)d_in[0];
    const int*   src_eids   = (const int*)d_in[1];
    const int*   dst_eids   = (const int*)d_in[2];
    const int*   src2_e1    = (const int*)d_in[3];
    const int*   src2_e2    = (const int*)d_in[4];
    const int*   dst2_e1    = (const int*)d_in[5];
    const int*   dst2_e2    = (const int*)d_in[6];
    const float* dt_src     = (const float*)d_in[7];
    const float* dt_dst     = (const float*)d_in[8];
    const float* dt2_src    = (const float*)d_in[9];
    const float* dt2_dst    = (const float*)d_in[10];
    const float* pcc1       = (const float*)d_in[11];
    const float* pcc2       = (const float*)d_in[12];
    const float* time_w     = (const float*)d_in[13];
    const float* time_b     = (const float*)d_in[14];
    const float* proj_W     = (const float*)d_in[15];
    const float* proj_b     = (const float*)d_in[16];
    const float* eproj_W    = (const float*)d_in[17];
    const float* eproj_b    = (const float*)d_in[18];
    const float* mix_tln_g  = (const float*)d_in[19];
    const float* mix_tln_b  = (const float*)d_in[20];
    const float* mix_tW1    = (const float*)d_in[21];
    const float* mix_tb1    = (const float*)d_in[22];
    const float* mix_tW2    = (const float*)d_in[23];
    const float* mix_tb2    = (const float*)d_in[24];
    const float* mix_cln_g  = (const float*)d_in[25];
    const float* mix_cln_b  = (const float*)d_in[26];
    const float* mix_cW1    = (const float*)d_in[27];
    const float* mix_cb1    = (const float*)d_in[28];
    const float* mix_cW2    = (const float*)d_in[29];
    const float* mix_cb2    = (const float*)d_in[30];

    char* ws = (char*)d_ws;
    float* tokens   = (float*)ws; ws += (size_t)NB * NTK * NC * sizeof(float);
    float* tokens2  = (float*)ws; ws += (size_t)NB * NTK * NC * sizeof(float);
    float* edge_emb = (float*)ws; ws += (size_t)NB * NC * sizeof(float);
    float* high_emb = (float*)ws; ws += (size_t)NB * NC * sizeof(float);
    _Float16* projWt  = (_Float16*)ws; ws += (size_t)NC * K1P * sizeof(_Float16);
    _Float16* eprojWt = (_Float16*)ws; ws += (size_t)2 * NC * K2P * sizeof(_Float16);
    _Float16* cW1t    = (_Float16*)ws; ws += (size_t)4 * NHC * NC * sizeof(_Float16);
    _Float16* cW2t    = (_Float16*)ws; ws += (size_t)4 * NC * NHC * sizeof(_Float16);

    // weight conversion to transposed padded f16
    wconv_kernel<<<(NC * K1P + 255) / 256, 256, 0, stream>>>(proj_W, projWt, ND + NT, NC, K1P);
    for (int j = 0; j < 2; ++j)
        wconv_kernel<<<(NC * K2P + 255) / 256, 256, 0, stream>>>(
            eproj_W + (size_t)j * (3 * ND + NT) * NC, eprojWt + (size_t)j * NC * K2P, 3 * ND + NT, NC, K2P);
    for (int i = 0; i < 4; ++i) {
        wconv_kernel<<<(NHC * NC + 255) / 256, 256, 0, stream>>>(
            mix_cW1 + (size_t)i * NC * NHC, cW1t + (size_t)i * NHC * NC, NC, NHC, NC);
        wconv_kernel<<<(NC * NHC + 255) / 256, 256, 0, stream>>>(
            mix_cW2 + (size_t)i * NHC * NC, cW2t + (size_t)i * NC * NHC, NHC, NC, NHC);
    }

    const int ntiles = (2 * HALF_ROWS) / 16;  // 5120
    build_tokens1<<<ntiles, 256, 0, stream>>>(edge_table, src_eids, dst_eids, dt_src, dt_dst,
                                              time_w, time_b, projWt, proj_b, tokens);
    build_tokens2<<<ntiles, 256, 0, stream>>>(edge_table, src_eids, dst_eids,
                                              src2_e1, src2_e2, dst2_e1, dst2_e2,
                                              dt2_src, dt2_dst, time_w, time_b,
                                              eprojWt, eproj_b, tokens2);

    mixer2_kernel<<<NB, 256, 0, stream>>>(tokens, edge_emb, 0,
                                          mix_tln_g, mix_tln_b, mix_tW1, mix_tb1, mix_tW2, mix_tb2,
                                          mix_cln_g, mix_cln_b, cW1t, mix_cb1, cW2t, mix_cb2);
    mixer2_kernel<<<NB, 256, 0, stream>>>(tokens2, high_emb, 2,
                                          mix_tln_g, mix_tln_b, mix_tW1, mix_tb1, mix_tW2, mix_tb2,
                                          mix_cln_g, mix_cln_b, cW1t, mix_cb1, cW2t, mix_cb2);

    combine_kernel<<<NB, NC, 0, stream>>>(edge_emb, high_emb, pcc1, pcc2, (float*)d_out);
}
